// Attention_26405458936467
// MI455X (gfx1250) — compile-verified, matrix operand order checked by reading, not run on hardware
//
#include <hip/hip_runtime.h>
#include <hip/hip_bf16.h>
#include <type_traits>

// ---------------------------------------------------------------------------
// Fused single-head attention for MI455X (gfx1250, wave32, WMMA).
//   out = softmax( (xWq^T+bq)(xWk^T+bk)^T / sqrt(A) ) (xWv^T+bv) Wo^T + bo
// All matmuls via v_wmma_f32_16x16x32_bf16 (f32 accumulate).
// Async LDS staging via global_load_async_to_lds_b128 (ASYNCcnt).
// ---------------------------------------------------------------------------

typedef __attribute__((ext_vector_type(16))) __bf16 bf16x16;
typedef __attribute__((ext_vector_type(8)))  __bf16 bf16x8;
typedef __attribute__((ext_vector_type(8)))  float  f32x8;
typedef __attribute__((ext_vector_type(4)))  float  f32x4;

#define B_SZ 4
#define SEQ  4096
#define DIM  1024

// 16x32 bf16 A/B fragment for v_wmma_f32_16x16x32_bf16, K-contiguous source.
// Lane layout (ISA 7.12.2): lanes 0-15 row r=lane, K {0..7, 16..23};
// lanes 16-31 row r=lane-16, K {8..15, 24..31}. Two 16B loads per lane.
__device__ __forceinline__ bf16x16 load_frag(const __bf16* base, int ld, int lane) {
  const int r  = lane & 15;
  const int kh = (lane & 16) ? 8 : 0;
  const __bf16* p = base + (size_t)r * (size_t)ld + kh;
  bf16x8 lo = *reinterpret_cast<const bf16x8*>(p);
  bf16x8 hi = *reinterpret_cast<const bf16x8*>(p + 16);
  return __builtin_shufflevector(lo, hi, 0,1,2,3,4,5,6,7,8,9,10,11,12,13,14,15);
}

__device__ __forceinline__ f32x8 wmma_bf16(bf16x16 a, bf16x16 b, f32x8 c) {
  return __builtin_amdgcn_wmma_f32_16x16x32_bf16(
      /*neg_a=*/false, a, /*neg_b=*/false, b,
      /*c_mod=*/(short)0, c, /*reuse_a=*/false, /*reuse_b=*/false);
}

// Async global->LDS 16B copy (VFLAT encoding: VDST = LDS byte address VGPR,
// VADDR = 64-bit global address). Flat->LDS translation truncates the generic
// shared pointer to its low 32 bits, so that is the DS address we pass.
__device__ __forceinline__ void async_copy_b128(void* lds_dst, const void* gsrc) {
  const unsigned int       l = (unsigned int)(uintptr_t)lds_dst;
  const unsigned long long g = (unsigned long long)(uintptr_t)gsrc;
  asm volatile("global_load_async_to_lds_b128 %0, %1, off"
               :: "v"(l), "v"(g) : "memory");
}
__device__ __forceinline__ void wait_async() {
  asm volatile("s_wait_asynccnt 0" ::: "memory");
}

// ---------------------------------------------------------------------------
// GEMM: C[M,N] = alpha * (A[M,K] * W[N,K]^T + bias[N])
// A row-major (K contiguous), W row-major (K contiguous)  -> both fragments
// load K-contiguous. fp32 operands converted to bf16 while staging to LDS;
// bf16 operands staged with async-to-LDS DMA.
// Block tile 128x128, 8 waves (2x4), wave tile 64x32, BK=32.
// TRANSB: store C transposed per batch as [b][col][s]   (used for V^T).
// ---------------------------------------------------------------------------
constexpr int BM = 128, BN = 128, BK = 32, LDT = BK + 8; // padded LDS stride

template <typename AT, typename OT, bool TRANSB>
__global__ __launch_bounds__(256) void gemm_wmma_kernel(
    const AT* __restrict__ A, const float* __restrict__ W,
    const float* __restrict__ bias, OT* __restrict__ C,
    int K, int N, float alpha, int seqLen) {
  __shared__ __attribute__((aligned(16))) __bf16 As[BM * LDT];
  __shared__ __attribute__((aligned(16))) __bf16 Bs[BN * LDT];

  const int tid  = threadIdx.x;
  const int lane = tid & 31;
  const int wave = tid >> 5;
  const int wm   = wave >> 2; // 0..1 -> 64-row group
  const int wn   = wave & 3;  // 0..3 -> 32-col group
  const size_t rowBase = (size_t)blockIdx.y * BM;
  const int    colBase = blockIdx.x * BN;

  f32x8 acc[4][2] = {};

  const int r0 = tid >> 2;       // 0..63
  const int c8 = (tid & 3) * 8;  // 0,8,16,24

  for (int kk = 0; kk < K; kk += BK) {
#pragma unroll
    for (int p = 0; p < 2; ++p) {
      const int r = r0 + p * 64;
      // stage A tile
      {
        const AT* s = A + (rowBase + r) * (size_t)K + kk + c8;
        if constexpr (std::is_same_v<AT, float>) {
          f32x4 a0 = *reinterpret_cast<const f32x4*>(s);
          f32x4 a1 = *reinterpret_cast<const f32x4*>(s + 4);
          bf16x8 t;
#pragma unroll
          for (int i = 0; i < 4; ++i) { t[i] = (__bf16)a0[i]; t[4 + i] = (__bf16)a1[i]; }
          *reinterpret_cast<bf16x8*>(&As[r * LDT + c8]) = t;
        } else {
          async_copy_b128(&As[r * LDT + c8], s); // bf16: straight DMA to LDS
        }
      }
      // stage W tile (fp32 -> bf16)
      {
        const float* w = W + (colBase + r) * (size_t)K + kk + c8;
        f32x4 b0 = *reinterpret_cast<const f32x4*>(w);
        f32x4 b1 = *reinterpret_cast<const f32x4*>(w + 4);
        bf16x8 t;
#pragma unroll
        for (int i = 0; i < 4; ++i) { t[i] = (__bf16)b0[i]; t[4 + i] = (__bf16)b1[i]; }
        *reinterpret_cast<bf16x8*>(&Bs[r * LDT + c8]) = t;
      }
    }
    if constexpr (!std::is_same_v<AT, float>) wait_async();
    __syncthreads();

    bf16x16 fa[4], fb[2];
#pragma unroll
    for (int mi = 0; mi < 4; ++mi)
      fa[mi] = load_frag(&As[(wm * 64 + mi * 16) * LDT], LDT, lane);
#pragma unroll
    for (int ni = 0; ni < 2; ++ni)
      fb[ni] = load_frag(&Bs[(wn * 32 + ni * 16) * LDT], LDT, lane);
#pragma unroll
    for (int mi = 0; mi < 4; ++mi)
#pragma unroll
      for (int ni = 0; ni < 2; ++ni)
        acc[mi][ni] = wmma_bf16(fa[mi], fb[ni], acc[mi][ni]);
    __syncthreads();
  }

  // Epilogue. C layout (ISA): VGPR e -> M = e (lanes 0-15) / e+8 (lanes 16-31),
  // N = lane%16.
  const int l15 = lane & 15;
  const int hi8 = (lane & 16) ? 8 : 0;
#pragma unroll
  for (int mi = 0; mi < 4; ++mi) {
#pragma unroll
    for (int ni = 0; ni < 2; ++ni) {
      const int col = colBase + wn * 32 + ni * 16 + l15;
      const float bv = bias[col];
#pragma unroll
      for (int e = 0; e < 8; ++e) {
        const size_t row = rowBase + wm * 64 + mi * 16 + e + hi8;
        const float val = alpha * (acc[mi][ni][e] + bv);
        if constexpr (TRANSB) {
          const size_t bi = row >> 12;            // / SEQ
          const size_t si = row & (size_t)(SEQ - 1);
          C[(bi * (size_t)N + col) * (size_t)seqLen + si] = (OT)val;
        } else {
          C[row * (size_t)N + col] = (OT)val;
        }
      }
    }
  }
}

// ---------------------------------------------------------------------------
// Flash attention: one block = (batch b, 32 query rows), 16 waves (512 thr).
// Per key iteration (128 keys): each wave computes one 16x16 score tile with
// 32 chained WMMAs over the 1024-dim reduction; softmax uses all 512 threads
// (16 lanes per row, shuffle reductions); ctx: each wave owns 64 output
// columns (64-VGPR accumulator) so all 4 V^T fragments stay live in registers
// and global loads pipeline with staggered s_wait_loadcnt.
// ---------------------------------------------------------------------------
constexpr int QT  = 32;        // query rows per block
constexpr int NK  = 128;       // keys per iteration
constexpr int HD  = 1024;      // attention dim
constexpr int QLD = HD + 8;    // LDS strides (16B-aligned, bank-staggered)
constexpr int SLD = NK + 4;
constexpr int PLD = NK + 8;

__global__ __launch_bounds__(512) void flash_attn_kernel(
    const __bf16* __restrict__ Q, const __bf16* __restrict__ Kb,
    const __bf16* __restrict__ Vt, __bf16* __restrict__ Ctx) {
  __shared__ __attribute__((aligned(16))) __bf16 Qs[QT * QLD];
  __shared__ __attribute__((aligned(16))) float  Ss[QT * SLD];
  __shared__ __attribute__((aligned(16))) __bf16 Ps[QT * PLD];
  __shared__ float mS[QT], lS[QT], corrS[QT];

  const int tid  = threadIdx.x;
  const int lane = tid & 31;
  const int wave = tid >> 5;        // 0..15
  const int b    = blockIdx.y;
  const size_t q0 = (size_t)blockIdx.x * QT;

  const __bf16* qg = Q  + ((size_t)b * SEQ + q0) * HD;
  const __bf16* kg = Kb + (size_t)b * SEQ * HD;
  const __bf16* vg = Vt + (size_t)b * HD * SEQ;

  // Stage the 32x1024 Q tile into LDS once via async DMA (16B per op).
  {
    const int r  = tid >> 4;          // 0..31
    const int c0 = (tid & 15) * 8;    // 0..120
    for (int c = c0; c < HD; c += 128)
      async_copy_b128(&Qs[r * QLD + c], &qg[(size_t)r * HD + c]);
  }
  if (tid < QT) { mS[tid] = -3.0e38f; lS[tid] = 0.0f; }
  wait_async();
  __syncthreads();

  const int sqr = (wave >> 3) * 16; // score q-row base (0/16)
  const int skc = (wave & 7) * 16;  // score key base (0..112)
  const int n0  = wave * 64;        // ctx column base (0..960)
  const int l15 = lane & 15;
  const int hi8 = (lane & 16) ? 8 : 0;

  // softmax mapping: 16 lanes per query row, 8 columns per lane
  const int srow = tid >> 4;
  const int scol = (tid & 15) * 8;

  f32x8 cacc[2][4] = {};

  for (int key0 = 0; key0 < SEQ; key0 += NK) {
    // warm L2 for the next key tile while we compute this one
    if (key0 + NK < SEQ) {
      const __bf16* nk = kg + (size_t)(key0 + NK + (tid >> 2)) * HD + (tid & 3) * 256;
      __builtin_prefetch(nk, 0, 1);
    }

    // ---- 1) scores tile (q already carries 1/sqrt(A) and bias) ----
    f32x8 sacc = {};
#pragma unroll 4
    for (int kk = 0; kk < HD; kk += 32) {
      bf16x16 fa = load_frag(&Qs[sqr * QLD + kk], QLD, lane);
      bf16x16 fb = load_frag(kg + (size_t)(key0 + skc) * HD + kk, HD, lane);
      sacc = wmma_bf16(fa, fb, sacc);
    }
#pragma unroll
    for (int e = 0; e < 8; ++e)
      Ss[(sqr + e + hi8) * SLD + skc + l15] = sacc[e];
    __syncthreads();

    // ---- 2) online softmax: 16 lanes per row, shuffle-reduced ----
    {
      const float mOld = mS[srow];
      float sv[8];
      float mNew = mOld;
#pragma unroll
      for (int j = 0; j < 8; ++j) {
        sv[j] = Ss[srow * SLD + scol + j];
        mNew  = fmaxf(mNew, sv[j]);
      }
#pragma unroll
      for (int off = 1; off < 16; off <<= 1)
        mNew = fmaxf(mNew, __shfl_xor(mNew, off, 16));
      float lsum = 0.0f;
#pragma unroll
      for (int j = 0; j < 8; ++j) {
        const float ev = __expf(sv[j] - mNew);
        Ps[srow * PLD + scol + j] = (__bf16)ev;
        lsum += ev;
      }
#pragma unroll
      for (int off = 1; off < 16; off <<= 1)
        lsum += __shfl_xor(lsum, off, 16);
      if ((tid & 15) == 0) {
        const float corr = __expf(mOld - mNew);
        lS[srow]    = lS[srow] * corr + lsum;
        mS[srow]    = mNew;
        corrS[srow] = corr;
      }
    }
    __syncthreads();

    // ---- 3) ctx += P @ V  (rescale old accumulator first) ----
#pragma unroll
    for (int mi = 0; mi < 2; ++mi)
#pragma unroll
      for (int e = 0; e < 8; ++e) {
        const float c = corrS[mi * 16 + e + hi8];
#pragma unroll
        for (int ni = 0; ni < 4; ++ni) cacc[mi][ni][e] *= c;
      }
#pragma unroll
    for (int ks = 0; ks < NK; ks += 32) {
      const bf16x16 fa0 = load_frag(&Ps[0 * PLD + ks], PLD, lane);
      const bf16x16 fa1 = load_frag(&Ps[16 * PLD + ks], PLD, lane);
      bf16x16 fbv[4];
#pragma unroll
      for (int ni = 0; ni < 4; ++ni)
        fbv[ni] = load_frag(vg + (size_t)(n0 + ni * 16) * SEQ + key0 + ks, SEQ, lane);
#pragma unroll
      for (int ni = 0; ni < 4; ++ni) {
        cacc[0][ni] = wmma_bf16(fa0, fbv[ni], cacc[0][ni]);
        cacc[1][ni] = wmma_bf16(fa1, fbv[ni], cacc[1][ni]);
      }
    }
    __syncthreads();
  }

  // ---- finalize: divide by l, store ctx as bf16 [b][s][a] ----
  __bf16* og = Ctx + ((size_t)b * SEQ + q0) * HD;
#pragma unroll
  for (int mi = 0; mi < 2; ++mi)
#pragma unroll
    for (int e = 0; e < 8; ++e) {
      const int row = mi * 16 + e + hi8;
      const float inv = 1.0f / lS[row];
#pragma unroll
      for (int ni = 0; ni < 4; ++ni) {
        const int col = n0 + ni * 16 + l15;
        og[(size_t)row * HD + col] = (__bf16)(cacc[mi][ni][e] * inv);
      }
    }
}

// ---------------------------------------------------------------------------
extern "C" void kernel_launch(void* const* d_in, const int* in_sizes, int n_in,
                              void* d_out, int out_size, void* d_ws, size_t ws_size,
                              hipStream_t stream) {
  (void)in_sizes; (void)n_in; (void)out_size; (void)ws_size;

  const float* x  = (const float*)d_in[0];
  const float* Wq = (const float*)d_in[1];
  const float* bq = (const float*)d_in[2];
  const float* Wk = (const float*)d_in[3];
  const float* bk = (const float*)d_in[4];
  const float* Wv = (const float*)d_in[5];
  const float* bv = (const float*)d_in[6];
  const float* Wo = (const float*)d_in[7];
  const float* bo = (const float*)d_in[8];
  float* out = (float*)d_out;

  // Workspace: q, k, v^T, ctx -- each B*S*A bf16 (32MB), 128MB total.
  const size_t tElems = (size_t)B_SZ * SEQ * DIM;
  __bf16* qws = (__bf16*)d_ws;
  __bf16* kws = qws + tElems;
  __bf16* vtw = kws + tElems;
  __bf16* ctx = vtw + tElems;

  dim3 blk(256);
  dim3 gproj(DIM / BN, (B_SZ * SEQ) / BM); // (8, 128)
  const float scale = 0.03125f;            // 1/sqrt(1024), folded into q

  gemm_wmma_kernel<float, __bf16, false>
      <<<gproj, blk, 0, stream>>>(x, Wq, bq, qws, DIM, DIM, scale, SEQ);
  gemm_wmma_kernel<float, __bf16, false>
      <<<gproj, blk, 0, stream>>>(x, Wk, bk, kws, DIM, DIM, 1.0f, SEQ);
  gemm_wmma_kernel<float, __bf16, true>
      <<<gproj, blk, 0, stream>>>(x, Wv, bv, vtw, DIM, DIM, 1.0f, SEQ);

  flash_attn_kernel<<<dim3(SEQ / QT, B_SZ), dim3(512), 0, stream>>>(qws, kws, vtw, ctx);

  gemm_wmma_kernel<__bf16, float, false>
      <<<gproj, blk, 0, stream>>>(ctx, Wo, bo, out, DIM, DIM, 1.0f, SEQ);
}
